// AttentionPooling_30227979829597
// MI455X (gfx1250) — matrix operand order repaired and checked by reading, not compiled, not run on hardware
//
#include <hip/hip_runtime.h>

// ---------------------------------------------------------------------------
// Attention pooling on gfx1250 (MI455X): bf16 WMMA + TDM (tensor_load_to_lds)
// K-panel staging, flash-style 2-pass softmax, column-sum trick so the P@V
// GEMM collapses into a weighted V reduction.
// ---------------------------------------------------------------------------

#define B_    4
#define S_    4096
#define E_    256
#define ROWS  (B_ * S_)
#define SCALE 0.0625f /* 1/sqrt(E) = 1/16 */
#define PANEL 32               /* K rows staged per LDS buffer (2 WMMA tiles) */
#define NPAN  (S_ / PANEL)     /* 128 panels */

typedef __attribute__((ext_vector_type(16))) __bf16 v16bf;
typedef __attribute__((ext_vector_type(8)))  __bf16 v8bf;
typedef __attribute__((ext_vector_type(8)))  float  v8f;
typedef __attribute__((ext_vector_type(4)))  unsigned int u32x4;
typedef __attribute__((ext_vector_type(8)))  int    i32x8;
typedef __attribute__((ext_vector_type(4)))  int    i32x4;

// A-operand (16x32 bf16, M x K): lane&15 = M row; lane half selects kb=0/16.
__device__ __forceinline__ v16bf load_a16x32(const __bf16* base, int koff, int lane) {
  const int r  = lane & 15;
  const int kb = (lane & 16) ? 16 : 0;
  const __bf16* p = base + r * E_ + koff + kb;
  v8bf lo = *(const v8bf*)(p);        // K = kb .. kb+7
  v8bf hi = *(const v8bf*)(p + 8);    // K = kb+8 .. kb+15
  return __builtin_shufflevector(lo, hi, 0,1,2,3,4,5,6,7,8,9,10,11,12,13,14,15);
}

// B-operand (32x16 bf16, K x N): lane&15 = N column; lanes 0-15 hold K=0..15,
// lanes 16-31 hold K=16..31 -> one contiguous 32-byte read per lane.
// Works from global or LDS (row-major rows of the transposed matrix).
__device__ __forceinline__ v16bf load_b32x16(const __bf16* base, int koff, int lane) {
  const int n  = lane & 15;
  const int ks = (lane & 16);         // 0 or 16
  return *(const v16bf*)(base + n * E_ + koff + ks);
}

__device__ __forceinline__ v8f wmma_bf16(v16bf a, v16bf b, v8f c) {
  return __builtin_amdgcn_wmma_f32_16x16x32_bf16(false, a, false, b, (short)0, c,
                                                 false, false);
}

// ---------------------------------------------------------------------------
// TDM: DMA a (rows x E_) bf16 row-major tile from global into contiguous LDS.
// D# group0: count=1, lds_addr, global_addr[56:0], type=2 ("image").
// D# group1: data_size=2B, tensor_dim0=E_, tensor_dim1=S_, tile_dim0=E_,
//            tile_dim1=rows, tensor_dim0_stride=E_. Groups 2/3 zero (2-D).
__device__ __forceinline__ void tdm_load_panel(unsigned lds_addr, const __bf16* gptr,
                                               unsigned rows) {
  const unsigned long long ga = (unsigned long long)(size_t)gptr;
  u32x4 g0;
  g0[0] = 1u;                                        // count=1, user mode
  g0[1] = lds_addr;                                  // lds_addr (bytes)
  g0[2] = (unsigned)(ga & 0xFFFFFFFFu);              // global_addr[31:0]
  g0[3] = (unsigned)((ga >> 32) & 0x01FFFFFFu)       // global_addr[56:32]
        | (2u << 30);                                // type=2
  i32x8 g1;
  g1[0] = (1 << 16);                                 // data_size=1 -> 2 bytes
  g1[1] = (int)((E_ & 0xFFFF) << 16);                // tensor_dim0[15:0] @ bit48
  g1[2] = (int)(((E_ >> 16) & 0xFFFF)                // tensor_dim0[31:16]
        | ((S_ & 0xFFFF) << 16));                    // tensor_dim1[15:0] @ bit80
  g1[3] = (int)(((S_ >> 16) & 0xFFFF)                // tensor_dim1[31:16]
        | ((E_ & 0xFFFF) << 16));                    // tile_dim0 @ bit112
  g1[4] = (int)(rows & 0xFFFF);                      // tile_dim1 @ bit128, tile_dim2=0
  g1[5] = E_;                                        // tensor_dim0_stride[31:0] @ bit160
  g1[6] = 0;                                         // stride0[47:32], stride1[15:0]
  g1[7] = 0;                                         // stride1[47:16]
  const i32x4 z4 = {0, 0, 0, 0};
  const i32x8 z8 = {0, 0, 0, 0, 0, 0, 0, 0};
  __builtin_amdgcn_tensor_load_to_lds(g0, g1, z4, z4, z8, 0);
}

// ---------------------------------------------------------------------------
// Kernel 0a: x (f32) -> xb (bf16)
__global__ __launch_bounds__(256) void k_convert_x(const float* __restrict__ x,
                                                   __bf16* __restrict__ xb) {
  int i = blockIdx.x * 256 + threadIdx.x;
  xb[i] = (__bf16)x[i];
}

// Kernel 0b: W[k][n] (f32) -> Wt[n][k] (bf16) for the three projections.
__global__ __launch_bounds__(256) void k_convert_wt(const float* __restrict__ Wq,
                                                    const float* __restrict__ Wk,
                                                    const float* __restrict__ Wv,
                                                    __bf16* __restrict__ wt_all) {
  const float* W = (blockIdx.y == 0) ? Wq : (blockIdx.y == 1) ? Wk : Wv;
  __bf16* Wt = wt_all + (size_t)blockIdx.y * E_ * E_;
  int i = blockIdx.x * 256 + threadIdx.x;  // i = n*E + k
  int n = i >> 8;
  int k = i & 255;
  Wt[i] = (__bf16)W[k * E_ + n];
}

// Kernel 0c: zero the per-batch column-weight accumulator (re-zeroed per launch)
__global__ __launch_bounds__(256) void k_zero_w(float* __restrict__ w) {
  w[blockIdx.x * 256 + threadIdx.x] = 0.0f;
}

// ---------------------------------------------------------------------------
// Kernel 1: QKV projection. grid = (ROWS/16, 3). 8 waves/block, 2 N-tiles/wave.
__global__ __launch_bounds__(256) void k_qkv_gemm(const __bf16* __restrict__ xb,
                                                  const __bf16* __restrict__ wt_all,
                                                  const float* __restrict__ bq,
                                                  const float* __restrict__ bk,
                                                  const float* __restrict__ bv,
                                                  __bf16* __restrict__ qkv_all) {
  const int lane  = threadIdx.x & 31;
  const int wave  = threadIdx.x >> 5;
  const int mtile = blockIdx.x;
  const int which = blockIdx.y;
  const __bf16* wt   = wt_all + (size_t)which * E_ * E_;
  const float*  bias = (which == 0) ? bq : (which == 1) ? bk : bv;
  __bf16*       out  = qkv_all + (size_t)which * ROWS * E_;
  const __bf16* abase = xb + (size_t)mtile * 16 * E_;

  v16bf a[8];
#pragma unroll
  for (int kc = 0; kc < 8; ++kc) a[kc] = load_a16x32(abase, kc * 32, lane);

#pragma unroll
  for (int ni = 0; ni < 2; ++ni) {
    const int nt = wave * 2 + ni;
    const float bval = bias[nt * 16 + (lane & 15)];
    v8f c;
#pragma unroll
    for (int i = 0; i < 8; ++i) c[i] = bval;  // bias: column fixed per lane
    const __bf16* bbase = wt + (size_t)nt * 16 * E_;
#pragma unroll
    for (int kc = 0; kc < 8; ++kc)
      c = wmma_bf16(a[kc], load_b32x16(bbase, kc * 32, lane), c);
    const int col   = nt * 16 + (lane & 15);
    const int rbase = mtile * 16 + ((lane & 16) ? 8 : 0);
#pragma unroll
    for (int v = 0; v < 8; ++v) out[(size_t)(rbase + v) * E_ + col] = (__bf16)c[v];
  }
}

// ---------------------------------------------------------------------------
// Kernel 2: pass 1 — per-row softmax max & sum. K panels staged by TDM into a
// double-buffered LDS tile shared by all 8 waves; per-lane online update.
__global__ __launch_bounds__(256) void k_attn_pass1(const __bf16* __restrict__ qb,
                                                    const __bf16* __restrict__ kbm,
                                                    float* __restrict__ Mrow,
                                                    float* __restrict__ Lrow) {
  __shared__ __bf16 kbuf[2][PANEL * E_];  // 2 x 16 KB

  const int lane = threadIdx.x & 31;
  const int wave = threadIdx.x >> 5;
  const int b    = blockIdx.y;
  const int rowbase = b * S_ + (blockIdx.x * 8 + wave) * 16;

  const __bf16* qbase = qb  + (size_t)rowbase * E_;
  const __bf16* kbase = kbm + (size_t)b * S_ * E_;

  v16bf a[8];
#pragma unroll
  for (int kc = 0; kc < 8; ++kc) a[kc] = load_a16x32(qbase, kc * 32, lane);

  float m[8], l[8];
#pragma unroll
  for (int v = 0; v < 8; ++v) { m[v] = -3.0e38f; l[v] = 0.0f; }

  if (wave == 0)  // one wave drives the DMA engine for the whole block
    tdm_load_panel((unsigned)(size_t)&kbuf[0][0], kbase, PANEL);

  for (int p = 0; p < NPAN; ++p) {
    const int buf = p & 1;
    if (wave == 0) {
      if (p + 1 < NPAN) {
        tdm_load_panel((unsigned)(size_t)&kbuf[buf ^ 1][0],
                       kbase + (size_t)(p + 1) * PANEL * E_, PANEL);
        __builtin_amdgcn_s_wait_tensorcnt(1);  // panel p landed, p+1 in flight
      } else {
        __builtin_amdgcn_s_wait_tensorcnt(0);
      }
    }
    __syncthreads();  // publish buf to all waves
#pragma unroll
    for (int tt = 0; tt < PANEL / 16; ++tt) {
      const __bf16* bbase = &kbuf[buf][tt * 16 * E_];
      v8f c;
#pragma unroll
      for (int i = 0; i < 8; ++i) c[i] = 0.0f;
#pragma unroll
      for (int kc = 0; kc < 8; ++kc)
        c = wmma_bf16(a[kc], load_b32x16(bbase, kc * 32, lane), c);
#pragma unroll
      for (int v = 0; v < 8; ++v) {
        const float s  = c[v] * SCALE;
        const float mn = fmaxf(m[v], s);
        l[v] = l[v] * __expf(m[v] - mn) + __expf(s - mn);
        m[v] = mn;
      }
    }
    __syncthreads();  // all waves done with buf before it is overwritten
  }

  // merge the 16 lanes that share each row (lanes differ only in column slot)
#pragma unroll
  for (int v = 0; v < 8; ++v) {
    float mf = m[v];
#pragma unroll
    for (int off = 1; off < 16; off <<= 1) mf = fmaxf(mf, __shfl_xor(mf, off, 32));
    float ls = l[v] * __expf(m[v] - mf);
#pragma unroll
    for (int off = 1; off < 16; off <<= 1) ls += __shfl_xor(ls, off, 32);
    if ((lane & 15) == 0) {
      const int r = rowbase + v + ((lane & 16) ? 8 : 0);
      Mrow[r] = mf;
      Lrow[r] = ls;
    }
  }
}

// ---------------------------------------------------------------------------
// Kernel 3: pass 2 — recompute scores, accumulate column weights w[t] = sum_s P.
__global__ __launch_bounds__(256) void k_attn_pass2(const __bf16* __restrict__ qb,
                                                    const __bf16* __restrict__ kbm,
                                                    const float* __restrict__ Mrow,
                                                    const float* __restrict__ Lrow,
                                                    float* __restrict__ wcol) {
  __shared__ __bf16 kbuf[2][PANEL * E_];  // 32 KB
  __shared__ float  lw[S_];               // 16 KB block-local column accumulator

  for (int i = threadIdx.x; i < S_; i += 256) lw[i] = 0.0f;

  const int lane = threadIdx.x & 31;
  const int wave = threadIdx.x >> 5;
  const int b    = blockIdx.y;
  const int rowbase = b * S_ + (blockIdx.x * 8 + wave) * 16;

  const __bf16* qbase = qb  + (size_t)rowbase * E_;
  const __bf16* kbase = kbm + (size_t)b * S_ * E_;

  v16bf a[8];
#pragma unroll
  for (int kc = 0; kc < 8; ++kc) a[kc] = load_a16x32(qbase, kc * 32, lane);

  float mr[8], il[8];
#pragma unroll
  for (int v = 0; v < 8; ++v) {
    const int r = rowbase + v + ((lane & 16) ? 8 : 0);
    mr[v] = Mrow[r];
    il[v] = 1.0f / Lrow[r];
  }

  if (wave == 0)
    tdm_load_panel((unsigned)(size_t)&kbuf[0][0], kbase, PANEL);

  for (int p = 0; p < NPAN; ++p) {
    const int buf = p & 1;
    if (wave == 0) {
      if (p + 1 < NPAN) {
        tdm_load_panel((unsigned)(size_t)&kbuf[buf ^ 1][0],
                       kbase + (size_t)(p + 1) * PANEL * E_, PANEL);
        __builtin_amdgcn_s_wait_tensorcnt(1);
      } else {
        __builtin_amdgcn_s_wait_tensorcnt(0);
      }
    }
    __syncthreads();
#pragma unroll
    for (int tt = 0; tt < PANEL / 16; ++tt) {
      const int t = p * (PANEL / 16) + tt;
      const __bf16* bbase = &kbuf[buf][tt * 16 * E_];
      v8f c;
#pragma unroll
      for (int i = 0; i < 8; ++i) c[i] = 0.0f;
#pragma unroll
      for (int kc = 0; kc < 8; ++kc)
        c = wmma_bf16(a[kc], load_b32x16(bbase, kc * 32, lane), c);
      float ps = 0.0f;  // this lane's 8 rows all share one column
#pragma unroll
      for (int v = 0; v < 8; ++v)
        ps += __expf(c[v] * SCALE - mr[v]) * il[v];
      ps += __shfl_xor(ps, 16, 32);  // pair lanes cover all 16 rows of the tile
      if (lane < 16) atomicAdd(&lw[t * 16 + lane], ps);  // ds_add_f32
    }
    __syncthreads();
  }

  for (int i = threadIdx.x; i < S_; i += 256)
    atomicAdd(&wcol[(size_t)b * S_ + i], lw[i]);  // global_atomic_add_f32
}

// ---------------------------------------------------------------------------
// Kernel 4: out[b,e] = (1/S) * sum_t w[b,t] * V[b,t,e]
__global__ __launch_bounds__(256) void k_finalize(const float* __restrict__ wcol,
                                                  const __bf16* __restrict__ vb,
                                                  float* __restrict__ out) {
  const int b = blockIdx.x;
  const int e = threadIdx.x;
  const __bf16* v = vb + (size_t)b * S_ * E_;
  const float*  w = wcol + (size_t)b * S_;
  float acc = 0.0f;
  for (int t = 0; t < S_; ++t) acc += w[t] * (float)v[(size_t)t * E_ + e];
  out[b * E_ + e] = acc * (1.0f / S_);
}

// ---------------------------------------------------------------------------
extern "C" void kernel_launch(void* const* d_in, const int* in_sizes, int n_in,
                              void* d_out, int out_size, void* d_ws, size_t ws_size,
                              hipStream_t stream) {
  const float* x  = (const float*)d_in[0];
  const float* Wq = (const float*)d_in[1];
  const float* bq = (const float*)d_in[2];
  const float* Wk = (const float*)d_in[3];
  const float* bk = (const float*)d_in[4];
  const float* Wv = (const float*)d_in[5];
  const float* bv = (const float*)d_in[6];
  float* out = (float*)d_out;

  char* ws = (char*)d_ws;
  size_t off = 0;
  auto alloc = [&](size_t bytes) {
    void* p = ws + off;
    off += (bytes + 255) & ~(size_t)255;
    return p;
  };
  __bf16* xb      = (__bf16*)alloc((size_t)ROWS * E_ * sizeof(__bf16));
  __bf16* wt_all  = (__bf16*)alloc((size_t)3 * E_ * E_ * sizeof(__bf16));
  __bf16* qkv_all = (__bf16*)alloc((size_t)3 * ROWS * E_ * sizeof(__bf16));
  float*  Mrow    = (float*)alloc((size_t)ROWS * sizeof(float));
  float*  Lrow    = (float*)alloc((size_t)ROWS * sizeof(float));
  float*  wcol    = (float*)alloc((size_t)ROWS * sizeof(float));
  __bf16* qb = qkv_all;
  __bf16* kb = qkv_all + (size_t)ROWS * E_;
  __bf16* vb = qkv_all + (size_t)2 * ROWS * E_;

  k_convert_x <<<dim3(ROWS * E_ / 256), 256, 0, stream>>>(x, xb);
  k_convert_wt<<<dim3(E_ * E_ / 256, 3), 256, 0, stream>>>(Wq, Wk, Wv, wt_all);
  k_zero_w    <<<dim3(ROWS / 256), 256, 0, stream>>>(wcol);
  k_qkv_gemm  <<<dim3(ROWS / 16, 3), 256, 0, stream>>>(xb, wt_all, bq, bk, bv, qkv_all);
  k_attn_pass1<<<dim3(S_ / 128, B_), 256, 0, stream>>>(qb, kb, Mrow, Lrow);
  k_attn_pass2<<<dim3(S_ / 128, B_), 256, 0, stream>>>(qb, kb, Mrow, Lrow, wcol);
  k_finalize  <<<4, 256, 0, stream>>>(wcol, vb, out);
}
